// GAttn_33062658245164
// MI455X (gfx1250) — compile-verified, matrix operand order checked by reading, not run on hardware
//
#include <hip/hip_runtime.h>
#include <hip/hip_bf16.h>

typedef __bf16 bf16;
typedef __attribute__((ext_vector_type(16))) __bf16 v16bf;
typedef __attribute__((ext_vector_type(8)))  __bf16 v8bf;
typedef __attribute__((ext_vector_type(4)))  __bf16 v4bf;
typedef __attribute__((ext_vector_type(8)))  float  v8f;
typedef unsigned int v4u __attribute__((ext_vector_type(4)));
typedef int v8i __attribute__((ext_vector_type(8)));
typedef int v4i __attribute__((ext_vector_type(4)));

#define B_ 8
#define H_ 16
#define L_ 1024
#define S_ 1024
#define D_ 1024
#define E_ 64

static __device__ __forceinline__ v8f zero8() {
  v8f z;
#pragma unroll
  for (int i = 0; i < 8; ++i) z[i] = 0.0f;
  return z;
}

// A/B fragment for v_wmma_f32_16x16x32_bf16 from a row-major bf16 matrix.
// Lane l -> row row0 + (l&15); K halves split by lane (l>>4) per ISA layout:
// VGPRs 0-3 hold K = half*8 .. +7, VGPRs 4-7 hold K = 16+half*8 .. +7.
static __device__ __forceinline__ v16bf frag_from_bf16(const bf16* __restrict__ base,
                                                       int row0, int kbase, int ld, int lane) {
  const int r = row0 + (lane & 15);
  const int half = lane >> 4;
  const bf16* p = base + (size_t)r * ld + kbase + half * 8;
  v8bf lo = *reinterpret_cast<const v8bf*>(p);
  v8bf hi = *reinterpret_cast<const v8bf*>(p + 16);
  v16bf f;
#pragma unroll
  for (int i = 0; i < 8; ++i) { f[i] = lo[i]; f[i + 8] = hi[i]; }
  return f;
}

static __device__ __forceinline__ v8f wmma_bf16(v16bf a, v16bf b, v8f c) {
  return __builtin_amdgcn_wmma_f32_16x16x32_bf16(false, a, false, b, (short)0, c, false, false);
}

// ---------------------------------------------------------------------------
// Vectorized f32 -> bf16 conversion (4 elements/thread; n % 1024 == 0).
// ---------------------------------------------------------------------------
__global__ __launch_bounds__(256) void cvt_f32_bf16_kernel(const float* __restrict__ in,
                                                           bf16* __restrict__ outp, size_t n) {
  size_t i = ((size_t)blockIdx.x * blockDim.x + threadIdx.x) * 4;
  if (i >= n) return;
  float4 f = *reinterpret_cast<const float4*>(in + i);
  v4bf o;
  o[0] = (bf16)f.x; o[1] = (bf16)f.y; o[2] = (bf16)f.z; o[3] = (bf16)f.w;
  *reinterpret_cast<v4bf*>(outp + i) = o;
}

// ---------------------------------------------------------------------------
// GEMM: out[m,n] = act( sum_k x[m,k] * W[n,k] + bias[n] ), x and W in bf16.
// One wave computes a 16x64 output tile (A fragment reused across 4 N-tiles).
// ---------------------------------------------------------------------------
template <bool RELU, bool OUT_BF16>
__global__ __launch_bounds__(256) void gemm_xWt_kernel(const bf16* __restrict__ x,
                                                       const bf16* __restrict__ W,
                                                       const float* __restrict__ bias,
                                                       void* __restrict__ outp,
                                                       int M, int N, int K) {
  const int lane  = threadIdx.x & 31;
  const int wave  = blockIdx.x * (blockDim.x >> 5) + (threadIdx.x >> 5);
  const int tilesN = N >> 6;
  const int m0 = (wave / tilesN) << 4;
  const int n0 = (wave % tilesN) << 6;
  if (m0 >= M) return;

  v8f acc[4];
#pragma unroll
  for (int t = 0; t < 4; ++t) acc[t] = zero8();

  for (int k0 = 0; k0 < K; k0 += 32) {
    if (k0 + 32 < K) {
      // warm next K-tile of the weight rows this lane will read
      __builtin_prefetch(W + (size_t)(n0 + (lane & 15)) * K + k0 + 32, 0, 0);
    }
    v16bf a = frag_from_bf16(x, m0, k0, K, lane);
#pragma unroll
    for (int t = 0; t < 4; ++t) {
      v16bf b = frag_from_bf16(W, n0 + t * 16, k0, K, lane);  // B[k][n] = W[n][k]
      acc[t] = wmma_bf16(a, b, acc[t]);
    }
  }

  const int half = lane >> 4, nn = lane & 15;
#pragma unroll
  for (int t = 0; t < 4; ++t) {
    const int n = n0 + t * 16 + nn;
    const float bv = bias[n];
#pragma unroll
    for (int i = 0; i < 8; ++i) {
      const int m = m0 + i + half * 8;
      float v = acc[t][i] + bv;
      if constexpr (RELU) v = v > 0.0f ? v : 0.0f;
      if constexpr (OUT_BF16) ((bf16*)outp)[(size_t)m * N + n] = (bf16)v;
      else                    ((float*)outp)[(size_t)m * N + n] = v;
    }
  }
}

// ---------------------------------------------------------------------------
// v [B,S,H,E] (bf16) -> vT [B,H,E,S] (bf16) so A@V B-fragments are K-contiguous
// ---------------------------------------------------------------------------
__global__ __launch_bounds__(256) void transpose_v_kernel(const bf16* __restrict__ vp,
                                                          bf16* __restrict__ vT) {
  size_t idx = (size_t)blockIdx.x * blockDim.x + threadIdx.x;
  if (idx >= (size_t)B_ * S_ * H_ * E_) return;
  const int e = idx & (E_ - 1);
  size_t t1 = idx >> 6;
  const int h = t1 & (H_ - 1);
  size_t t2 = t1 >> 4;
  const int s = t2 & (S_ - 1);
  const int b = (int)(t2 >> 10);
  vT[(((size_t)b * H_ + h) * E_ + e) * S_ + s] = vp[idx];
}

// ---------------------------------------------------------------------------
// Fully fused attention core for one (b, h, 16 query rows) block:
//   phase 1: Q tile -> LDS via Tensor Data Mover
//   phase 2: scores = q k^T via WMMA -> 16x1024 f32 tile in LDS
//   phase 3: per-row top-8 threshold + masked softmax; write f32 A to d_out
//            and bf16 A tile to LDS (A never re-read from HBM)
//   phase 4: A @ V via WMMA (A frags from LDS, V from L2-resident vT),
//            8 per-wave K-slice partials reduced through LDS -> attnV
// ---------------------------------------------------------------------------
__global__ __launch_bounds__(256) void attn_fused_kernel(const bf16* __restrict__ qp,
                                                         const bf16* __restrict__ kp,
                                                         const bf16* __restrict__ vT,
                                                         float* __restrict__ A,
                                                         bf16* __restrict__ attnV) {
  __shared__ bf16  sq[16 * E_];   // 2 KB Q tile
  __shared__ float sc[16 * S_];   // 64 KB score tile; reused as reduction buffer
  __shared__ bf16  sa[16 * S_];   // 32 KB bf16 A tile

  const int lane = threadIdx.x & 31;
  const int wave = threadIdx.x >> 5;  // 0..7 -> columns wave*128 .. +127
  const int blk  = blockIdx.x;        // (b*H + h)*(L/16) + ltile
  const int ltile = blk & (L_ / 16 - 1);
  const int bh    = blk >> 6;
  const int h = bh & (H_ - 1);
  const int b = bh >> 4;
  const int l0 = ltile * 16;

  const bf16* qtile = qp + (((size_t)b * L_ + l0) * H_ + h) * E_;   // row l, ld = H*E
  const bf16* kbase = kp + ((size_t)b * S_ * H_ + h) * E_;          // row s, ld = H*E

  // ---- phase 1: stage the 16x64 Q tile (row stride H*E) into LDS via TDM ----
#if __has_builtin(__builtin_amdgcn_tensor_load_to_lds) && __has_builtin(__builtin_amdgcn_s_wait_tensorcnt)
  if (wave == 0) {
    const unsigned lds_addr = (unsigned)(uintptr_t)(void*)sq;
    const unsigned long long ga = (unsigned long long)(uintptr_t)qtile;
    v4u g0;
    g0[0] = 1u;                                   // count=1, user descriptor
    g0[1] = lds_addr;                             // lds_addr[31:0]
    g0[2] = (unsigned)ga;                         // global_addr[31:0]
    g0[3] = ((unsigned)(ga >> 32) & 0x01FFFFFFu)  // global_addr[56:32]
            | 0x80000000u;                        // type=2 ("image")
    v8i g1;
    g1[0] = 0x00010000;                           // data_size=1 (2 bytes)
    g1[1] = (int)((unsigned)E_ << 16);            // tensor_dim0 = 64 (lo16 at bit48)
    g1[2] = (int)(16u << 16);                     // tensor_dim0 hi=0 | tensor_dim1 lo=16
    g1[3] = (int)((unsigned)E_ << 16);            // tensor_dim1 hi=0 | tile_dim0 = 64
    g1[4] = 16;                                   // tile_dim1 = 16 | tile_dim2 = 0
    g1[5] = H_ * E_;                              // tensor_dim0_stride = 1024 (lo32)
    g1[6] = 0;
    g1[7] = 0;
    v4i z4; z4[0] = z4[1] = z4[2] = z4[3] = 0;
    v8i z8;
#pragma unroll
    for (int i = 0; i < 8; ++i) z8[i] = 0;
    __builtin_amdgcn_tensor_load_to_lds(g0, g1, z4, z4, z8, 0);
    __builtin_amdgcn_s_wait_tensorcnt(0);
  }
  __syncthreads();
#else
  for (int i = threadIdx.x; i < 16 * E_ / 4; i += blockDim.x) {
    const int r = (i * 4) / E_, c = (i * 4) & (E_ - 1);
    *reinterpret_cast<v4bf*>(&sq[r * E_ + c]) =
        *reinterpret_cast<const v4bf*>(qtile + (size_t)r * (H_ * E_) + c);
  }
  __syncthreads();
#endif

  // ---- phase 2: scores tile -> sc ----
  {
    v8f acc[8];
#pragma unroll
    for (int t = 0; t < 8; ++t) acc[t] = zero8();
#pragma unroll
    for (int e0 = 0; e0 < E_; e0 += 32) {
      v16bf a = frag_from_bf16(sq, 0, e0, E_, lane);  // Q frags from LDS
#pragma unroll
      for (int t = 0; t < 8; ++t) {
        v16bf kb = frag_from_bf16(kbase, wave * 128 + t * 16, e0, H_ * E_, lane);
        acc[t] = wmma_bf16(a, kb, acc[t]);
      }
    }
    const int half = lane >> 4, nn = lane & 15;
#pragma unroll
    for (int t = 0; t < 8; ++t)
#pragma unroll
      for (int i = 0; i < 8; ++i)
        sc[(i + 8 * half) * S_ + wave * 128 + t * 16 + nn] = acc[t][i];
  }
  __syncthreads();

  // ---- phase 3: top-8 threshold + masked softmax; A (f32) out, sa (bf16) in LDS ----
  float* Arow = A + (((size_t)bh) * L_ + l0) * S_;
  for (int rr = 0; rr < 2; ++rr) {
    const int row = wave * 2 + rr;
    float vals[32];
#pragma unroll
    for (int j = 0; j < 32; ++j) vals[j] = sc[row * S_ + lane + 32 * j];

    float thresh = __builtin_inff();
    float rowmax = -__builtin_inff();
#pragma unroll
    for (int it = 0; it < 8; ++it) {
      float m = -__builtin_inff();
#pragma unroll
      for (int j = 0; j < 32; ++j) {
        const float v = vals[j];
        if (v < thresh) m = fmaxf(m, v);
      }
#pragma unroll
      for (int off = 16; off > 0; off >>= 1) m = fmaxf(m, __shfl_xor(m, off, 32));
      if (it == 0) rowmax = m;
      thresh = m;
    }

    float ex[32];
    float sum = 0.0f;
#pragma unroll
    for (int j = 0; j < 32; ++j) {
      const float v = vals[j];
      const float e = (v >= thresh) ? __expf(0.125f * (v - rowmax)) : 0.0f;
      ex[j] = e;
      sum += e;
    }
#pragma unroll
    for (int off = 16; off > 0; off >>= 1) sum += __shfl_xor(sum, off, 32);
    const float rinv = 1.0f / sum;
#pragma unroll
    for (int j = 0; j < 32; ++j) {
      const float a = ex[j] * rinv;
      Arow[(size_t)row * S_ + lane + 32 * j] = a;   // required f32 output
      sa[row * S_ + lane + 32 * j] = (bf16)a;       // on-chip copy for A@V
    }
  }
  __syncthreads();

  // ---- phase 4: A @ V, per-wave K-slice partials, reduce through LDS ----
  {
    const bf16* vb = vT + (size_t)bh * E_ * S_;     // row e, ld = S
    v8f acc[4];
#pragma unroll
    for (int t = 0; t < 4; ++t) acc[t] = zero8();
#pragma unroll
    for (int ks = 0; ks < 4; ++ks) {
      const int s0 = wave * 128 + ks * 32;
      v16bf a = frag_from_bf16(sa, 0, s0, S_, lane);
#pragma unroll
      for (int t = 0; t < 4; ++t) {
        v16bf bb = frag_from_bf16(vb, t * 16, s0, S_, lane);  // B[k=s][n=e] = vT[e][s]
        acc[t] = wmma_bf16(a, bb, acc[t]);
      }
    }
    // partials into the (dead) score buffer: red[w][t][i][lane]
    float* red = sc;
#pragma unroll
    for (int t = 0; t < 4; ++t)
#pragma unroll
      for (int i = 0; i < 8; ++i)
        red[wave * 1024 + (t * 8 + i) * 32 + lane] = acc[t][i];
  }
  __syncthreads();

  // reduce 8 partials; each thread produces 4 of the 16x64 outputs
  {
    const int tid = threadIdx.x;
#pragma unroll
    for (int q = 0; q < 4; ++q) {
      const int p = tid * 4 + q;
      float s = 0.0f;
#pragma unroll
      for (int w = 0; w < 8; ++w) s += sc[w * 1024 + p];
      const int t  = p >> 8;
      const int i  = (p >> 5) & 7;
      const int l2 = p & 31;
      const int row = i + 8 * (l2 >> 4);
      const int col = t * 16 + (l2 & 15);
      attnV[(((size_t)b * L_ + l0 + row) * H_ + h) * E_ + col] = (bf16)s;
    }
  }
}

// ---------------------------------------------------------------------------

extern "C" void kernel_launch(void* const* d_in, const int* in_sizes, int n_in,
                              void* d_out, int out_size, void* d_ws, size_t ws_size,
                              hipStream_t stream) {
  const float* queries = (const float*)d_in[0];
  const float* keys    = (const float*)d_in[1];
  const float* values  = (const float*)d_in[2];
  const float* Wq = (const float*)d_in[3];
  const float* bq = (const float*)d_in[4];
  const float* Wk = (const float*)d_in[5];
  const float* bk = (const float*)d_in[6];
  const float* Wv = (const float*)d_in[7];
  const float* bv = (const float*)d_in[8];
  const float* Wo = (const float*)d_in[9];
  const float* bo = (const float*)d_in[10];

  float* out = (float*)d_out;                       // [B,L,D]
  float* A   = out + (size_t)B_ * L_ * D_;          // [B,H,L,S]

  const size_t MD = (size_t)B_ * L_ * D_;           // 8.39M elements
  const size_t WD = (size_t)D_ * D_;                // 1.05M elements
  bf16* xq    = (bf16*)d_ws;                        // bf16 activations
  bf16* xk    = xq + MD;
  bf16* xv    = xk + MD;
  bf16* wqb   = xv + MD;                            // bf16 weights
  bf16* wkb   = wqb + WD;
  bf16* wvb   = wkb + WD;
  bf16* wob   = wvb + WD;
  bf16* qp    = wob + WD;                           // projected q/k/v
  bf16* kp    = qp + MD;
  bf16* vp    = kp + MD;
  bf16* vT    = vp + MD;
  bf16* attnV = vT + MD;

  const int M = B_ * L_;  // 8192
  const int gemmBlocks = (M / 16) * (D_ / 64) / 8;  // 1024 blocks of 8 waves
  const int cvtBlkAct = (int)(MD / 4 / 256);        // 8192
  const int cvtBlkW   = (int)(WD / 4 / 256);        // 1024

  // Pre-convert activations and weights to bf16 (one pass each).
  cvt_f32_bf16_kernel<<<cvtBlkAct, 256, 0, stream>>>(queries, xq, MD);
  cvt_f32_bf16_kernel<<<cvtBlkAct, 256, 0, stream>>>(keys,    xk, MD);
  cvt_f32_bf16_kernel<<<cvtBlkAct, 256, 0, stream>>>(values,  xv, MD);
  cvt_f32_bf16_kernel<<<cvtBlkW,   256, 0, stream>>>(Wq, wqb, WD);
  cvt_f32_bf16_kernel<<<cvtBlkW,   256, 0, stream>>>(Wk, wkb, WD);
  cvt_f32_bf16_kernel<<<cvtBlkW,   256, 0, stream>>>(Wv, wvb, WD);
  cvt_f32_bf16_kernel<<<cvtBlkW,   256, 0, stream>>>(Wo, wob, WD);

  // Q/K/V projections (relu on Q and V)
  gemm_xWt_kernel<true,  true><<<gemmBlocks, 256, 0, stream>>>(xq, wqb, bq, qp, M, D_, D_);
  gemm_xWt_kernel<false, true><<<gemmBlocks, 256, 0, stream>>>(xk, wkb, bk, kp, M, D_, D_);
  gemm_xWt_kernel<true,  true><<<gemmBlocks, 256, 0, stream>>>(xv, wvb, bv, vp, M, D_, D_);

  // v -> vT for K-contiguous B fragments in A@V
  transpose_v_kernel<<<(int)(MD / 256), 256, 0, stream>>>(vp, vT);

  // scores + top-k + softmax + A@V, fully fused (A written once, never re-read)
  attn_fused_kernel<<<B_ * H_ * (L_ / 16), 256, 0, stream>>>(qp, kp, vT, A, attnV);

  // output projection, bf16 in -> f32 out
  gemm_xWt_kernel<false, false><<<gemmBlocks, 256, 0, stream>>>(attnV, wob, bo, out, M, D_, D_);
}
